// BGNNLayer_9010841387249
// MI455X (gfx1250) — matrix operand order repaired and checked by reading, not compiled, not run on hardware
//
#include <hip/hip_runtime.h>
#include <hip/hip_bf16.h>

typedef unsigned short u16;
typedef unsigned int   u32;
typedef __attribute__((ext_vector_type(16))) __bf16 v16bf;
typedef __attribute__((ext_vector_type(8)))  float  v8f;

// Problem constants (fixed by the reference)
#define POLY_DIM 128
#define PAIR_DIM 128
#define HIDDEN   256
#define IN_DIM   384          // PAIR_DIM + 2*POLY_DIM
#define NPOLY    4096
#define NPAIR    65536
#define BATCH    4
#define M_BLOCK  128          // pairs per block (8 m-tiles)
#define LN_EPS   1e-5f

// ---- fp32 -> bf16 (round to nearest even) ----
__device__ __forceinline__ u16 f2bf(float f) {
    u32 u = __float_as_uint(f);
    u32 r = u + 0x7FFFu + ((u >> 16) & 1u);
    return (u16)(r >> 16);
}

// load a 16-element bf16 fragment as two 16B chunks
__device__ __forceinline__ v16bf ld_frag(const u16* __restrict__ base, int off0, int off1) {
    union { v16bf v; uint4 q[2]; } f;
    f.q[0] = *reinterpret_cast<const uint4*>(base + off0);
    f.q[1] = *reinterpret_cast<const uint4*>(base + off1);
    return f.v;
}

// Accumulate two 16-row strips x 8 adjacent 16x16 tiles:
//   C[u][16, n0..n0+127] += A[arow0 + 16u .., :] * Wt^T     (u = 0,1)
// A in LDS (row stride ASTRIDE bf16); Wt in global, N-major bf16: Wt[n*KDIM + k].
// Each B fragment is loaded once and feeds TWO WMMAs (both M-strips).
template<int KDIM, int ASTRIDE>
__device__ __forceinline__ void gemm_acc(v8f acc[2][8],
                                         const u16* __restrict__ Abase, int arow0,
                                         const u16* __restrict__ Wt, int n0,
                                         int ml, int hi) {
    const int aoff = hi * 8;    // A: lanes 16-31 start at K+8
    const int boff = hi * 8;    // B: lanes 16-31 start at K+8
    const u16* arow0p = Abase + (size_t)(arow0 + ml) * ASTRIDE;
    const u16* arow1p = Abase + (size_t)(arow0 + 16 + ml) * ASTRIDE;
    for (int ks = 0; ks < KDIM / 32; ++ks) {
        const int k0 = ks * 32;
        v16bf a0 = ld_frag(arow0p, k0 + aoff, k0 + aoff + 16);
        v16bf a1 = ld_frag(arow1p, k0 + aoff, k0 + aoff + 16);
#pragma unroll
        for (int t = 0; t < 8; ++t) {
            const u16* brow = Wt + (size_t)(n0 + t * 16 + ml) * KDIM;
            v16bf bf = ld_frag(brow, k0 + boff, k0 + boff + 16);
            acc[0][t] = __builtin_amdgcn_wmma_f32_16x16x32_bf16(
                false, a0, false, bf, (short)0, acc[0][t], false, false);
            acc[1][t] = __builtin_amdgcn_wmma_f32_16x16x32_bf16(
                false, a1, false, bf, (short)0, acc[1][t], false, false);
        }
    }
}

// ---- prep: convert W1/W2/Wu (f32, K-major) to bf16 N-major in workspace ----
__global__ void bgnn_prep_weights(const float* __restrict__ W1,
                                  const float* __restrict__ W2,
                                  const float* __restrict__ Wu,
                                  u16* __restrict__ ws) {
    int i = blockIdx.x * blockDim.x + threadIdx.x;
    const int n1 = HIDDEN * IN_DIM;        // 98304
    const int n2 = n1 + HIDDEN * HIDDEN;   // 163840
    const int n3 = n2 + HIDDEN * HIDDEN;   // 229376
    if (i < n1) {
        int n = i / IN_DIM, k = i - n * IN_DIM;
        ws[i] = f2bf(W1[(size_t)k * HIDDEN + n]);
    } else if (i < n2) {
        int j = i - n1; int n = j / HIDDEN, k = j - n * HIDDEN;
        ws[i] = f2bf(W2[(size_t)k * HIDDEN + n]);
    } else if (i < n3) {
        int j = i - n2; int n = j / HIDDEN, k = j - n * HIDDEN;
        ws[i] = f2bf(Wu[(size_t)k * HIDDEN + n]);
    }
}

// ---- main fused kernel: gather -> MLP(2) -> LN -> update, 128 pairs / block ----
__global__ __launch_bounds__(256) void bgnn_layer_kernel(
    const float* __restrict__ poly,   // (B, NPOLY, 128)
    const float* __restrict__ pair,   // (B, NPAIR, 128)
    const int*   __restrict__ pidx,   // (B, NPAIR, 2)
    const float* __restrict__ b1,
    const float* __restrict__ b2,
    const float* __restrict__ lng,
    const float* __restrict__ lnb,
    const float* __restrict__ bu,
    const u16*   __restrict__ W1t,    // 256 x 384 bf16
    const u16*   __restrict__ W2t,    // 256 x 256 bf16
    const u16*   __restrict__ Wut,    // 256 x 256 bf16
    float* __restrict__ out)          // (B, NPAIR, 256)
{
    // dynamic LDS: 128 x 384 bf16 = 96 KB (combined A; reused for h / messages)
    extern __shared__ __align__(16) u16 sBuf[];
    __shared__ float sSum[M_BLOCK];
    __shared__ float sSumSq[M_BLOCK];
    u16* sH = sBuf;                         // M_BLOCK x 256 view (64 KB)

    const int tid  = threadIdx.x;
    const int lane = tid & 31;
    const int wid  = tid >> 5;       // 0..7
    const int ml   = lane & 15;
    const int hi   = lane >> 4;
    const int mg   = wid & 3;        // M-group: 32 rows (two 16-row tiles)
    const int nh   = wid >> 2;       // N-half (128 cols each)
    const int arow0 = mg * 32;
    const int n0    = nh * 128;

    if (tid < M_BLOCK) { sSum[tid] = 0.f; sSumSq[tid] = 0.f; }

    // -------- Phase 0: gather parents, concat, convert to bf16 in LDS --------
    // 128 rows x 384 cols = 128 x 96 float4 chunks; 48 chunks per thread.
#pragma unroll 4
    for (int it = 0; it < (M_BLOCK * IN_DIM / 4) / 256; ++it) {
        int c = tid + it * 256;
        int r = c / 96;
        int col4 = (c - r * 96) * 4;
        size_t grow = (size_t)blockIdx.x * M_BLOCK + r;
        int b = (int)(grow / NPAIR);
        int p = (int)(grow - (size_t)b * NPAIR);
        const float* src;
        if (col4 < PAIR_DIM) {
            src = pair + ((size_t)b * NPAIR + p) * PAIR_DIM + col4;
        } else {
            int which = (col4 < PAIR_DIM + POLY_DIM) ? 0 : 1;
            int idx = pidx[((size_t)b * NPAIR + p) * 2 + which];
            src = poly + ((size_t)b * NPOLY + idx) * POLY_DIM
                       + (col4 - PAIR_DIM - which * POLY_DIM);
        }
        float4 f = *reinterpret_cast<const float4*>(src);
        uint2 pk;
        pk.x = (u32)f2bf(f.x) | ((u32)f2bf(f.y) << 16);
        pk.y = (u32)f2bf(f.z) | ((u32)f2bf(f.w) << 16);
        *reinterpret_cast<uint2*>(&sBuf[(size_t)r * IN_DIM + col4]) = pk;
    }
    __syncthreads();

    // -------- Stage 1: h = relu(combined @ W1 + b1) --------
    v8f acc[2][8] = {};
    gemm_acc<IN_DIM, IN_DIM>(acc, sBuf, arow0, W1t, n0, ml, hi);
    __syncthreads();   // everyone done reading combined A; reuse buffer for h
#pragma unroll
    for (int u = 0; u < 2; ++u) {
#pragma unroll
        for (int t = 0; t < 8; ++t) {
            const int n = n0 + t * 16 + ml;
            const float bias = b1[n];
#pragma unroll
            for (int r = 0; r < 8; ++r) {
                float v = acc[u][t][r] + bias;
                v = fmaxf(v, 0.f);
                sH[(size_t)(arow0 + u * 16 + hi * 8 + r) * HIDDEN + n] = f2bf(v);
            }
        }
    }
    __syncthreads();

    // -------- Stage 2: h2 = h @ W2 + b2 --------
#pragma unroll
    for (int u = 0; u < 2; ++u)
#pragma unroll
        for (int t = 0; t < 8; ++t) acc[u][t] = (v8f)(0.f);
    gemm_acc<HIDDEN, HIDDEN>(acc, sH, arow0, W2t, n0, ml, hi);
#pragma unroll
    for (int u = 0; u < 2; ++u)
#pragma unroll
        for (int t = 0; t < 8; ++t) {
            const float bias = b2[n0 + t * 16 + ml];
#pragma unroll
            for (int r = 0; r < 8; ++r) acc[u][t][r] += bias;
        }

    // -------- LayerNorm reductions (Σ, Σ²) via LDS f32 atomics --------
#pragma unroll
    for (int u = 0; u < 2; ++u)
#pragma unroll
        for (int r = 0; r < 8; ++r) {
            float s = 0.f, q = 0.f;
#pragma unroll
            for (int t = 0; t < 8; ++t) { float v = acc[u][t][r]; s += v; q += v * v; }
            const int m = arow0 + u * 16 + hi * 8 + r;
            atomicAdd(&sSum[m], s);
            atomicAdd(&sSumSq[m], q);
        }
    __syncthreads();

    float mu[2][8], rstd[2][8];
#pragma unroll
    for (int u = 0; u < 2; ++u)
#pragma unroll
        for (int r = 0; r < 8; ++r) {
            const int m = arow0 + u * 16 + hi * 8 + r;
            mu[u][r] = sSum[m] * (1.f / HIDDEN);
            float var = sSumSq[m] * (1.f / HIDDEN) - mu[u][r] * mu[u][r];
            rstd[u][r] = rsqrtf(var + LN_EPS);
        }
#pragma unroll
    for (int t = 0; t < 8; ++t) {
        const int n = n0 + t * 16 + ml;
        const float g = lng[n], bb = lnb[n];
#pragma unroll
        for (int u = 0; u < 2; ++u)
#pragma unroll
            for (int r = 0; r < 8; ++r) {
                float msg = (acc[u][t][r] - mu[u][r]) * rstd[u][r] * g + bb;
                sH[(size_t)(arow0 + u * 16 + hi * 8 + r) * HIDDEN + n] = f2bf(msg);
            }
    }
    __syncthreads();

    // -------- Stage 3: out = messages @ Wu + bu --------
#pragma unroll
    for (int u = 0; u < 2; ++u)
#pragma unroll
        for (int t = 0; t < 8; ++t) acc[u][t] = (v8f)(0.f);
    gemm_acc<HIDDEN, HIDDEN>(acc, sH, arow0, Wut, n0, ml, hi);

#pragma unroll
    for (int u = 0; u < 2; ++u) {
        const size_t rowbase = (size_t)blockIdx.x * M_BLOCK + arow0 + u * 16 + hi * 8;
#pragma unroll
        for (int t = 0; t < 8; ++t) {
            const int n = n0 + t * 16 + ml;
            const float bias = bu[n];
#pragma unroll
            for (int r = 0; r < 8; ++r) {
                out[(rowbase + r) * HIDDEN + n] = acc[u][t][r] + bias;
            }
        }
    }
}

extern "C" void kernel_launch(void* const* d_in, const int* in_sizes, int n_in,
                              void* d_out, int out_size, void* d_ws, size_t ws_size,
                              hipStream_t stream) {
    (void)in_sizes; (void)n_in; (void)out_size; (void)ws_size;
    const float* poly = (const float*)d_in[0];
    const float* pair = (const float*)d_in[1];
    const int*   pidx = (const int*)d_in[2];
    const float* W1   = (const float*)d_in[3];
    const float* b1   = (const float*)d_in[4];
    const float* W2   = (const float*)d_in[5];
    const float* b2   = (const float*)d_in[6];
    const float* lng  = (const float*)d_in[7];
    const float* lnb  = (const float*)d_in[8];
    const float* Wu   = (const float*)d_in[9];
    const float* bu   = (const float*)d_in[10];

    u16* ws  = (u16*)d_ws;
    u16* W1t = ws;                                    // 256*384
    u16* W2t = ws + HIDDEN * IN_DIM;                  // 256*256
    u16* Wut = ws + HIDDEN * IN_DIM + HIDDEN * HIDDEN;

    const int prep_total = HIDDEN * IN_DIM + 2 * HIDDEN * HIDDEN;   // 229376
    bgnn_prep_weights<<<(prep_total + 255) / 256, 256, 0, stream>>>(W1, W2, Wu, ws);

    const int nblocks = (BATCH * NPAIR) / M_BLOCK;    // 2048
    const size_t lds_bytes = (size_t)M_BLOCK * IN_DIM * sizeof(u16);   // 96 KB dynamic LDS
    bgnn_layer_kernel<<<nblocks, 256, lds_bytes, stream>>>(
        poly, pair, pidx, b1, b2, lng, lnb, bu, W1t, W2t, Wut, (float*)d_out);
}
